// DAInsHead_13391708028973
// MI455X (gfx1250) — matrix operand order repaired and checked by reading, not compiled, not run on hardware
//
#include <hip/hip_runtime.h>

#define DD 1024
#define HH 1024
#define LL 4

typedef __attribute__((ext_vector_type(16))) __bf16 v16bf;
typedef __attribute__((ext_vector_type(8)))  __bf16 v8bf;
typedef __attribute__((ext_vector_type(8)))  float  v8f;

static __device__ __forceinline__ unsigned short f2bf(float f) {
    union { float f; unsigned u; } v; v.f = f;
    unsigned r = v.u + 0x7FFFu + ((v.u >> 16) & 1u);   // round-to-nearest-even
    return (unsigned short)(r >> 16);
}

// ---- prep: cast + transpose weights to bf16, [L][K][N] -> [L][N][K] ----
__global__ void cast_transpose_w(const float* __restrict__ W,
                                 unsigned short* __restrict__ Wt,
                                 int K, int Nc) {
    long long e = (long long)blockIdx.x * blockDim.x + threadIdx.x;
    long long total = (long long)LL * K * Nc;
    if (e >= total) return;
    int per = K * Nc;
    int l   = (int)(e / per);
    int rem = (int)(e % per);
    int n = rem / K;
    int k = rem % K;
    Wt[e] = f2bf(W[(long long)l * per + (long long)k * Nc + n]);
}

// ---- bucket rows by level: meta[0..3]=counts [4..7]=offsets [8..11]=cursors ----
__global__ void zero_meta(int* meta) { if (threadIdx.x < 12) meta[threadIdx.x] = 0; }

__global__ void count_levels(const int* __restrict__ levels, int* meta, int n) {
    int i = blockIdx.x * blockDim.x + threadIdx.x;
    if (i < n) atomicAdd(&meta[levels[i] & 3], 1);
}

__global__ void scan_levels(int* meta) {
    if (threadIdx.x == 0) {
        int o = 0;
        for (int l = 0; l < 4; ++l) { meta[4 + l] = o; meta[8 + l] = o; o += meta[l]; }
    }
}

__global__ void scatter_levels(const int* __restrict__ levels, int* meta,
                               int* __restrict__ idxbuf, int n) {
    int i = blockIdx.x * blockDim.x + threadIdx.x;
    if (i < n) { int p = atomicAdd(&meta[8 + (levels[i] & 3)], 1); idxbuf[p] = i; }
}

// ---- fused routed MLP: 64 rows/block, 8 waves (each owns 64x16 slab) ----
// LDS: Abuf 64x1024 bf16 (128K) + Hbuf 64x1024 bf16 (128K) + outsum + rowidx
__global__ __launch_bounds__(256)
void mlp_fused(const float* __restrict__ x,
               const unsigned short* __restrict__ w1t,
               const unsigned short* __restrict__ w2t,
               const float* __restrict__ b1,
               const float* __restrict__ b2,
               const float* __restrict__ W3,
               const float* __restrict__ b3,
               const int* __restrict__ meta,
               const int* __restrict__ idxbuf,
               float* __restrict__ out) {
    extern __shared__ unsigned short smem[];
    unsigned short* Abuf = smem;                       // [64][1024] bf16 (x tile)
    unsigned short* Hbuf = smem + 64 * HH;             // [64][1024] bf16 (h1)
    float* outsum = (float*)(smem + 128 * HH);         // [64]
    int*   rowidx = (int*)(outsum + 64);               // [64]

    const int lvl  = blockIdx.y;
    const int cnt  = meta[lvl];
    const int tile = blockIdx.x * 64;
    if (tile >= cnt) return;                           // uniform exit
    const int off   = meta[4 + lvl];
    const int valid = min(64, cnt - tile);

    const int tid  = threadIdx.x;
    const int lane = tid & 31;
    const int wv   = tid >> 5;          // 0..7 : each wave owns a 16-col slab
    const int ln   = lane & 15;
    const int hi   = (lane >> 4) & 1;   // half-wave

    if (tid < 64) {
        outsum[tid] = 0.0f;
        int rloc = tid < valid ? tid : (valid - 1);    // clamped padding
        rowidx[tid] = idxbuf[off + tile + rloc];
    }
    __syncthreads();

    // ---- stage gathered x rows into LDS as bf16 (one conversion/element) ----
    for (int e = tid; e < 64 * (DD / 4); e += 256) {
        const int row = e >> 8;                        // DD/4 == 256 float4 per row
        const int c4  = e & 255;
        const float4 v4 = ((const float4*)(x + (size_t)rowidx[row] * DD))[c4];
        uint2 p;
        p.x = (unsigned)f2bf(v4.x) | ((unsigned)f2bf(v4.y) << 16);
        p.y = (unsigned)f2bf(v4.z) | ((unsigned)f2bf(v4.w) << 16);
        ((uint2*)Abuf)[(row << 8) + c4] = p;           // ds_store_b64
    }
    __syncthreads();

    const unsigned short* w1l = w1t + (size_t)lvl * HH * DD;
    const unsigned short* w2l = w2t + (size_t)lvl * HH * HH;
    const float* b1l = b1 + lvl * HH;
    const float* b2l = b2 + lvl * HH;
    const float* w3l = W3 + lvl * HH;

    const int ka = hi ? 8 : 0;          // A-frag half-wave K shift (elements)
    const int kb = hi ? 8 : 0;          // B-frag half-wave K shift (elements)

    // ---------- Layer 1: h1 = relu(x @ W1 + b1) -> Hbuf (bf16) ----------
    for (int nc = 0; nc < HH; nc += 128) {
        const int n = nc + wv * 16 + ln;               // this wave's column
        v8f acc[4] = {};
        for (int k0 = 0; k0 < DD; k0 += 32) {
            const v16bf b = __builtin_shufflevector(*(const v8bf*)(w1l + (size_t)n * DD + k0 + kb), *(const v8bf*)(w1l + (size_t)n * DD + k0 + kb + 16), 0,1,2,3,4,5,6,7,8,9,10,11,12,13,14,15);
#pragma unroll
            for (int t = 0; t < 4; ++t) {              // 4 M-tiles of 16 rows
                const unsigned short* ap = Abuf + (size_t)(t * 16 + ln) * DD + k0 + ka;
                v8bf alo = *(const v8bf*)(ap);
                v8bf ahi = *(const v8bf*)(ap + 16);
                v16bf a = __builtin_shufflevector(alo, ahi,
                    0,1,2,3,4,5,6,7,8,9,10,11,12,13,14,15);
                acc[t] = __builtin_amdgcn_wmma_f32_16x16x32_bf16(
                    false, a, false, b, (short)0, acc[t], false, false);
            }
        }
        const float bias = b1l[n];
#pragma unroll
        for (int t = 0; t < 4; ++t)
#pragma unroll
            for (int i = 0; i < 8; ++i) {
                float v = acc[t][i] + bias;
                v = v > 0.0f ? v : 0.0f;
                Hbuf[(size_t)(t * 16 + i + hi * 8) * HH + n] = f2bf(v);
            }
    }
    __syncthreads();

    // ------- Layer 2 (+ fused layer 3 dot): relu(h1 @ W2 + b2) . w3 -------
    float acc3[4][8] = {};
    for (int nc = 0; nc < HH; nc += 128) {
        const int n = nc + wv * 16 + ln;
        v8f acc[4] = {};
        for (int k0 = 0; k0 < HH; k0 += 32) {
            const v16bf b = __builtin_shufflevector(*(const v8bf*)(w2l + (size_t)n * HH + k0 + kb), *(const v8bf*)(w2l + (size_t)n * HH + k0 + kb + 16), 0,1,2,3,4,5,6,7,8,9,10,11,12,13,14,15);
#pragma unroll
            for (int t = 0; t < 4; ++t) {
                const unsigned short* ap = Hbuf + (size_t)(t * 16 + ln) * HH + k0 + ka;
                v8bf alo = *(const v8bf*)(ap);
                v8bf ahi = *(const v8bf*)(ap + 16);
                v16bf a = __builtin_shufflevector(alo, ahi,
                    0,1,2,3,4,5,6,7,8,9,10,11,12,13,14,15);
                acc[t] = __builtin_amdgcn_wmma_f32_16x16x32_bf16(
                    false, a, false, b, (short)0, acc[t], false, false);
            }
        }
        const float bias = b2l[n];
        const float w3v  = w3l[n];
#pragma unroll
        for (int t = 0; t < 4; ++t)
#pragma unroll
            for (int i = 0; i < 8; ++i) {
                float v = acc[t][i] + bias;
                v = v > 0.0f ? v : 0.0f;
                acc3[t][i] += v * w3v;
            }
    }
    // reduce each row-partial over the 16 lanes of its half-wave (sum over N)
#pragma unroll
    for (int t = 0; t < 4; ++t)
#pragma unroll
        for (int i = 0; i < 8; ++i) {
            float s = acc3[t][i];
            s += __shfl_xor(s, 1);
            s += __shfl_xor(s, 2);
            s += __shfl_xor(s, 4);
            s += __shfl_xor(s, 8);
            if (ln == 0) atomicAdd(&outsum[t * 16 + i + hi * 8], s); // ds_add_f32
        }
    __syncthreads();

    if (tid < 64 && tid < valid) {
        out[rowidx[tid]] = outsum[tid] + b3[lvl];
    }
}

extern "C" void kernel_launch(void* const* d_in, const int* in_sizes, int n_in,
                              void* d_out, int out_size, void* d_ws, size_t ws_size,
                              hipStream_t stream) {
    const float* x      = (const float*)d_in[0];
    const int*   levels = (const int*)d_in[1];
    const float* W1     = (const float*)d_in[2];
    const float* b1     = (const float*)d_in[3];
    const float* W2     = (const float*)d_in[4];
    const float* b2     = (const float*)d_in[5];
    const float* W3     = (const float*)d_in[6];
    const float* b3     = (const float*)d_in[7];
    float* out = (float*)d_out;
    const int N = in_sizes[1];

    // workspace carve-up (~18 MB)
    size_t o = 0;
    unsigned short* w1t = (unsigned short*)((char*)d_ws + o); o += (size_t)LL * HH * DD * 2;
    unsigned short* w2t = (unsigned short*)((char*)d_ws + o); o += (size_t)LL * HH * HH * 2;
    int* idxbuf = (int*)((char*)d_ws + o); o += (size_t)N * 4;
    int* meta   = (int*)((char*)d_ws + o); o += 64;
    (void)ws_size; (void)n_in; (void)out_size;

    {
        long long tot = (long long)LL * DD * HH;
        int thr = 256;
        unsigned blk = (unsigned)((tot + thr - 1) / thr);
        cast_transpose_w<<<blk, thr, 0, stream>>>(W1, w1t, DD, HH);
        cast_transpose_w<<<blk, thr, 0, stream>>>(W2, w2t, HH, HH);
    }
    zero_meta<<<1, 32, 0, stream>>>(meta);
    count_levels<<<(N + 255) / 256, 256, 0, stream>>>(levels, meta, N);
    scan_levels<<<1, 32, 0, stream>>>(meta);
    scatter_levels<<<(N + 255) / 256, 256, 0, stream>>>(levels, meta, idxbuf, N);

    dim3 grid((N + 63) / 64, LL);
    // 128K (x tile) + 128K (h1) + 64 f32 + 64 i32 of dynamic LDS (<= 320K/WGP)
    size_t shmem = (size_t)128 * HH * 2 + 64 * sizeof(float) + 64 * sizeof(int);
    mlp_fused<<<grid, 256, shmem, stream>>>(x, w1t, w2t, b1, b2, W3, b3,
                                            meta, idxbuf, out);
}
